// SingleHeadAttention_71889162600745
// MI455X (gfx1250) — matrix operand order repaired and checked by reading, not compiled, not run on hardware
//
#include <hip/hip_runtime.h>
#include <math.h>

#define D 128
#define NEG_SLOPE 0.2f

typedef __attribute__((ext_vector_type(16))) _Float16 v16h;
typedef __attribute__((ext_vector_type(8)))  _Float16 v8h;
typedef __attribute__((ext_vector_type(8)))  float    v8f;

// ---------------------------------------------------------------------------
// Kernel 0: zero/init scratch: q_agg=0, out=0, node_max=-inf, node_sum=0
// ---------------------------------------------------------------------------
__global__ void gat_init(float* __restrict__ qagg, float* __restrict__ out,
                         float* __restrict__ nmax, float* __restrict__ nsum,
                         int n_nodes) {
    int idx = blockIdx.x * blockDim.x + threadIdx.x;
    int tot = n_nodes * D;
    if (idx < tot) { qagg[idx] = 0.f; out[idx] = 0.f; }
    if (idx < n_nodes) { nmax[idx] = -INFINITY; nsum[idx] = 0.f; }
}

// ---------------------------------------------------------------------------
// Kernel 1: fused triple GEMM with WMMA f16 (fp32 accumulate), FULL tiles only.
//   h_q   = x @ W_q
//   k_act = leaky_relu(x @ W_k)
//   h_proj= x @ W_l + b_l
// One wave computes a 16-row node tile; W matrices staged in LDS transposed
// to [n][k] f16 so B-fragments are contiguous 16B LDS loads. All stores are
// unconditional (remainder rows handled by gat_gemm3_tail).
// ---------------------------------------------------------------------------
__global__ __launch_bounds__(128) void gat_gemm3(
        const float* __restrict__ x,
        const float* __restrict__ Wq, const float* __restrict__ Wk,
        const float* __restrict__ Wl, const float* __restrict__ bl,
        float* __restrict__ hq, float* __restrict__ kact,
        float* __restrict__ hproj, int numFullTiles) {
    extern __shared__ _Float16 smem[];   // 3 * 128*128 f16, layout [m][n][k]

    const int tid = threadIdx.x;
    // ---- stage W_q, W_k, W_l into LDS, transposed + converted to f16 ----
    const float* Ws[3] = { Wq, Wk, Wl };
    #pragma unroll
    for (int m = 0; m < 3; ++m) {
        const float* W = Ws[m];
        for (int idx = tid; idx < D * D; idx += blockDim.x) {
            int k = idx >> 7;          // source row    (in-dim)
            int n = idx & (D - 1);     // source column (out-dim) -> coalesced
            smem[m * D * D + n * D + k] = (_Float16)W[idx];
        }
    }
    __syncthreads();

    const int lane = tid & 31;
    const int wave = tid >> 5;
    const int r16  = lane & 15;
    const int hi   = lane >> 4;                 // lane half selects K half
    const int gw     = blockIdx.x * (blockDim.x >> 5) + wave;
    const int nWaves = gridDim.x * (blockDim.x >> 5);

    // per-lane bias values for the 8 column tiles (hoisted out of tile loop)
    float biasv[8];
    #pragma unroll
    for (int nt = 0; nt < 8; ++nt) biasv[nt] = bl[nt * 16 + r16];

    for (int tile = gw; tile < numFullTiles; tile += nWaves) {
        const int rowBase = tile * 16;

        // ---- build the 4 A-fragments (16x32 f16 each) for K = 0..127 ----
        const float* px = x + (size_t)(rowBase + r16) * D;
        v16h afr[4];
        #pragma unroll
        for (int kc = 0; kc < 4; ++kc) {
            const int k0 = kc * 32 + hi * 16;
            float4 f0 = *(const float4*)(px + k0);
            float4 f1 = *(const float4*)(px + k0 + 4);
            float4 f2 = *(const float4*)(px + k0 + 8);
            float4 f3 = *(const float4*)(px + k0 + 12);
            v16h a;
            a[0] = (_Float16)f0.x;  a[1] = (_Float16)f0.y;
            a[2] = (_Float16)f0.z;  a[3] = (_Float16)f0.w;
            a[4] = (_Float16)f1.x;  a[5] = (_Float16)f1.y;
            a[6] = (_Float16)f1.z;  a[7] = (_Float16)f1.w;
            a[8] = (_Float16)f2.x;  a[9] = (_Float16)f2.y;
            a[10] = (_Float16)f2.z; a[11] = (_Float16)f2.w;
            a[12] = (_Float16)f3.x; a[13] = (_Float16)f3.y;
            a[14] = (_Float16)f3.z; a[15] = (_Float16)f3.w;
            afr[kc] = a;
        }

        // ---- 3 outputs x 8 column tiles, K=128 via 4 WMMAs each ----
        #pragma unroll
        for (int m = 0; m < 3; ++m) {
            float* dst = (m == 0) ? hq : (m == 1) ? kact : hproj;
            for (int nt = 0; nt < 8; ++nt) {
                const int ncol = nt * 16 + r16;        // output column
                v8f c = {};
                #pragma unroll
                for (int kc = 0; kc < 4; ++kc) {
                    const v8h* bp = (const v8h*)(smem + m * D * D +
                                                 ncol * D + kc * 32 + hi * 16);
                    v8h blo = bp[0];
                    v8h bhi = bp[1];
                    v16h b = __builtin_shufflevector(blo, bhi,
                        0, 1, 2, 3, 4, 5, 6, 7, 8, 9, 10, 11, 12, 13, 14, 15);
                    c = __builtin_amdgcn_wmma_f32_16x16x32_f16(
                            false, afr[kc], false, b, (short)0, c,
                            false, false);
                }
                // epilogue: leaky-relu for k_act, +bias for h_proj
                #pragma unroll
                for (int rr = 0; rr < 8; ++rr) {
                    float y = c[rr];
                    if (m == 1) y = (y > 0.f) ? y : NEG_SLOPE * y;
                    if (m == 2) y += biasv[nt];
                    c[rr] = y;
                }
                // unconditional stores: lanes 0-15 write row (rowBase+rr),
                // lanes 16-31 write row (rowBase+8+rr), 64B contiguous each
                float* pout = dst + (size_t)(rowBase + hi * 8) * D + ncol;
                #pragma unroll
                for (int rr = 0; rr < 8; ++rr)
                    pout[(size_t)rr * D] = c[rr];
            }
        }
    }
}

// ---------------------------------------------------------------------------
// Kernel 1b: scalar tail for the last (n_nodes % 16) rows. One thread per
// (row, col) output element; only launched when a remainder exists.
// ---------------------------------------------------------------------------
__global__ void gat_gemm3_tail(
        const float* __restrict__ x,
        const float* __restrict__ Wq, const float* __restrict__ Wk,
        const float* __restrict__ Wl, const float* __restrict__ bl,
        float* __restrict__ hq, float* __restrict__ kact,
        float* __restrict__ hproj, int rowStart, int n_nodes) {
    int idx = blockIdx.x * blockDim.x + threadIdx.x;
    int row = rowStart + idx / D;
    int col = idx % D;
    if (row >= n_nodes) return;
    const float* px = x + (size_t)row * D;
    float sq = 0.f, sk = 0.f, sl = 0.f;
    for (int k = 0; k < D; ++k) {
        float xv = px[k];
        sq += xv * Wq[k * D + col];
        sk += xv * Wk[k * D + col];
        sl += xv * Wl[k * D + col];
    }
    size_t o = (size_t)row * D + col;
    hq[o]    = sq;
    kact[o]  = (sk > 0.f) ? sk : NEG_SLOPE * sk;
    hproj[o] = sl + bl[col];
}

// ---------------------------------------------------------------------------
// Kernel 2: q_agg[row[e]] += h_q[col[e]]  (wave per edge, 4 floats per lane)
// ---------------------------------------------------------------------------
__global__ void gat_qagg(const int* __restrict__ row, const int* __restrict__ col,
                         const float* __restrict__ hq, float* __restrict__ qagg,
                         int nE) {
    int e = (blockIdx.x * blockDim.x + threadIdx.x) >> 5;
    int lane = threadIdx.x & 31;
    if (e >= nE) return;
    int r = row[e], cn = col[e];
    float4 v = *(const float4*)(hq + (size_t)cn * D + lane * 4);
    float* p = qagg + (size_t)r * D + lane * 4;
    atomicAdd(p + 0, v.x); atomicAdd(p + 1, v.y);
    atomicAdd(p + 2, v.z); atomicAdd(p + 3, v.w);
}

// ---------------------------------------------------------------------------
// Kernel 3: scores[e] = dot(k_act[row[e]], q_agg[col[e]]) / sqrt(D)
//           node_max[row[e]] = max(...)  (wave per edge + wave reduction)
// ---------------------------------------------------------------------------
__global__ void gat_scores(const int* __restrict__ row, const int* __restrict__ col,
                           const float* __restrict__ kact,
                           const float* __restrict__ qagg,
                           float* __restrict__ scores, float* __restrict__ nmax,
                           int nE) {
    int e = (blockIdx.x * blockDim.x + threadIdx.x) >> 5;
    int lane = threadIdx.x & 31;
    if (e >= nE) return;
    int r = row[e], cn = col[e];
    float4 a = *(const float4*)(kact + (size_t)r * D + lane * 4);
    float4 b = *(const float4*)(qagg + (size_t)cn * D + lane * 4);
    float s = a.x * b.x + a.y * b.y + a.z * b.z + a.w * b.w;
    #pragma unroll
    for (int off = 16; off > 0; off >>= 1)
        s += __shfl_xor(s, off, 32);
    s *= 0.08838834764831845f;            // 1/sqrt(128)
    if (lane == 0) {
        scores[e] = s;
        atomicMax(nmax + r, s);           // global_atomic_max_num_f32
    }
}

// ---------------------------------------------------------------------------
// Kernel 4: exp_scores[e] = exp(scores[e] - max[row[e]]); sum per row
// ---------------------------------------------------------------------------
__global__ void gat_expsum(const int* __restrict__ row,
                           float* __restrict__ scores,
                           const float* __restrict__ nmax,
                           float* __restrict__ nsum, int nE) {
    int e = blockIdx.x * blockDim.x + threadIdx.x;
    if (e >= nE) return;
    int r = row[e];
    float ex = __expf(scores[e] - nmax[r]);
    scores[e] = ex;
    atomicAdd(nsum + r, ex);
}

// ---------------------------------------------------------------------------
// Kernel 5: out[row[e]] += alpha[e] * h_proj[col[e]]  (wave per edge)
// ---------------------------------------------------------------------------
__global__ void gat_aggregate(const int* __restrict__ row,
                              const int* __restrict__ col,
                              const float* __restrict__ escore,
                              const float* __restrict__ nsum,
                              const float* __restrict__ hproj,
                              float* __restrict__ out, int nE) {
    int e = (blockIdx.x * blockDim.x + threadIdx.x) >> 5;
    int lane = threadIdx.x & 31;
    if (e >= nE) return;
    int r = row[e], cn = col[e];
    float alpha = escore[e] / (nsum[r] + 1e-8f);
    float4 v = *(const float4*)(hproj + (size_t)cn * D + lane * 4);
    float* p = out + (size_t)r * D + lane * 4;
    atomicAdd(p + 0, alpha * v.x); atomicAdd(p + 1, alpha * v.y);
    atomicAdd(p + 2, alpha * v.z); atomicAdd(p + 3, alpha * v.w);
}

// ---------------------------------------------------------------------------
// Kernel 6: final leaky_relu in place on out
// ---------------------------------------------------------------------------
__global__ void gat_final(float* __restrict__ out, int tot) {
    int idx = blockIdx.x * blockDim.x + threadIdx.x;
    if (idx < tot) {
        float v = out[idx];
        out[idx] = (v > 0.f) ? v : NEG_SLOPE * v;
    }
}

// ---------------------------------------------------------------------------
extern "C" void kernel_launch(void* const* d_in, const int* in_sizes, int n_in,
                              void* d_out, int out_size, void* d_ws, size_t ws_size,
                              hipStream_t stream) {
    const float* x  = (const float*)d_in[0];
    const int*   ei = (const int*)d_in[1];
    const float* Wq = (const float*)d_in[2];
    const float* Wk = (const float*)d_in[3];
    // d_in[4] = W_v : computed-but-unused in the reference -> skipped entirely
    const float* Wl = (const float*)d_in[5];
    const float* bl = (const float*)d_in[6];

    const int n_nodes = in_sizes[0] / D;
    const int nE      = in_sizes[1] / 2;
    const int* row = ei;            // edge_index[0]
    const int* col = ei + nE;       // edge_index[1]

    float* ws = (float*)d_ws;
    const size_t Nd = (size_t)n_nodes * D;
    float* hq     = ws;
    float* kact   = ws + Nd;
    float* hproj  = ws + 2 * Nd;
    float* qagg   = ws + 3 * Nd;
    float* scores = ws + 4 * Nd;
    float* nmax   = scores + nE;
    float* nsum   = nmax + n_nodes;
    float* out    = (float*)d_out;

    const int tot = n_nodes * D;

    gat_init<<<(tot + 255) / 256, 256, 0, stream>>>(qagg, out, nmax, nsum, n_nodes);

    const int numFullTiles = n_nodes / 16;          // WMMA path: full tiles only
    const int tailRows     = n_nodes - numFullTiles * 16;
    if (numFullTiles > 0) {
        const int tilesPerWave = 4;
        const int gemmBlocks =
            (numFullTiles + 4 * tilesPerWave - 1) / (4 * tilesPerWave);
        const size_t ldsBytes = 3 * D * D * sizeof(_Float16);   // 96 KB
        gat_gemm3<<<gemmBlocks, 128, ldsBytes, stream>>>(
            x, Wq, Wk, Wl, bl, hq, kact, hproj, numFullTiles);
    }
    if (tailRows > 0) {
        const int tailElems = tailRows * D;
        gat_gemm3_tail<<<(tailElems + 127) / 128, 128, 0, stream>>>(
            x, Wq, Wk, Wl, bl, hq, kact, hproj, numFullTiles * 16, n_nodes);
    }

    const int eWaveBlocks = (nE + 7) / 8;      // 256 threads = 8 waves/block
    gat_qagg<<<eWaveBlocks, 256, 0, stream>>>(row, col, hq, qagg, nE);
    gat_scores<<<eWaveBlocks, 256, 0, stream>>>(row, col, kact, qagg, scores, nmax, nE);
    gat_expsum<<<(nE + 255) / 256, 256, 0, stream>>>(row, scores, nmax, nsum, nE);
    gat_aggregate<<<eWaveBlocks, 256, 0, stream>>>(row, col, scores, nsum, hproj, out, nE);
    gat_final<<<(tot + 255) / 256, 256, 0, stream>>>(out, tot);
}
